// TSARNet_58823872086497
// MI455X (gfx1250) — compile-verified, matrix operand order checked by reading, not run on hardware
//
#include <hip/hip_runtime.h>
#include <hip/hip_bf16.h>

// ---------------------------------------------------------------------------
// TSARNet layer (only layer L-1 matters: h is overwritten each layer and the
// graph message path never consumes h).  Pipeline:
//   P0: pack node features / Wm[2] / Wl[2] to bf16 (uint-pair storage)
//   P1: agg = boundary_condition (init)
//   P2: edge kernel: msg = relu(bf16-WMMA([nf[src]|attr] @ Wm^T) + bm);
//       atomicAdd into agg[dst]
//   P3: node kernel: out = bf16-WMMA(agg @ Wl^T) + bl; LayerNorm; relu
// ---------------------------------------------------------------------------

#define NN   50000
#define NE   800000
#define EMB  128
#define ATTR 16
#define KDIM 144          // EMB + ATTR
#define LSEL 2            // only the last layer's result survives

typedef __attribute__((ext_vector_type(16))) __bf16    v16bf;
typedef __attribute__((ext_vector_type(8)))  float     v8f;
typedef __attribute__((ext_vector_type(8)))  unsigned  v8u;

static __device__ __forceinline__ unsigned f2bf(float f) {
    unsigned u = __float_as_uint(f);
    return (u + 0x7FFFu + ((u >> 16) & 1u)) >> 16;        // RNE
}
static __device__ __forceinline__ unsigned pack2(float lo, float hi) {
    return f2bf(lo) | (f2bf(hi) << 16);
}
static __device__ __forceinline__ v16bf as_bf(v8u u) {
    union { v8u a; v16bf b; } c; c.a = u; return c.b;
}
// per-lane K offset within a 32-wide K tile (ISA 16-bit A/B fragment layout)
static __device__ __forceinline__ int koff(int j, int half) {
    return (j < 4) ? (half * 8 + 2 * j) : (16 + half * 8 + 2 * (j - 4));
}

// ---------------------------------------------------------------- prep ----
__global__ void pack_bf16_pairs(const float* __restrict__ src,
                                unsigned* __restrict__ dst, int npairs) {
    int i = blockIdx.x * blockDim.x + threadIdx.x;
    if (i < npairs) dst[i] = pack2(src[2 * i], src[2 * i + 1]);
}

__global__ void init_agg(const float* __restrict__ bc,
                         float* __restrict__ agg, int n) {
    int i = blockIdx.x * blockDim.x + threadIdx.x;
    if (i < n) agg[i] = bc[i];
}

// ------------------------------------------------- edge GEMM + scatter ----
// block = 256 threads = 8 waves; wave -> tile of 16 edges x 128 outputs.
__global__ __launch_bounds__(256) void edge_gemm_scatter(
    const unsigned* __restrict__ nfb,        // [NN,128] bf16 as 64 uints/row
    const unsigned* __restrict__ wmb,        // [128,144] bf16 as 72 uints/row
    const float*    __restrict__ edge_attr,  // [NE,16]
    const int*      __restrict__ edge_index, // [2,NE]
    const float*    __restrict__ bm2,        // [128]
    float*          __restrict__ agg)        // [NN,128]
{
    __shared__ unsigned wlds[9216];          // 36 KB: Wm[2] in bf16
    for (int i = threadIdx.x; i < 9216; i += 256) wlds[i] = wmb[i];
    __syncthreads();

    const int lane = threadIdx.x & 31;
    const int wave = threadIdx.x >> 5;
    const int half = lane >> 4;
    const int lm   = lane & 15;
    const int tile = blockIdx.x * 8 + wave;
    if (tile >= NE / 16) return;
    const int e0 = tile * 16;

    const int src = edge_index[e0 + lm];     // A-row gather index (row = lm)
    int dstr[8];
    #pragma unroll
    for (int r = 0; r < 8; ++r) dstr[r] = edge_index[NE + e0 + half * 8 + r];

    v8f acc[8];
    #pragma unroll
    for (int nt = 0; nt < 8; ++nt)
        #pragma unroll
        for (int i = 0; i < 8; ++i) acc[nt][i] = 0.0f;

    #pragma unroll
    for (int kt = 0; kt < 5; ++kt) {         // K = 144 -> 5 tiles of 32 (padded)
        const int kb = kt * 32;
        v8u au;
        #pragma unroll
        for (int j = 0; j < 8; ++j) {
            const int cj = (j < 4) ? 2 * j : (16 + 2 * (j - 4));
            const int k  = kb + koff(j, half);
            if (kb + cj < EMB) {                       // node-feature region
                au[j] = nfb[(size_t)src * 64 + (k >> 1)];
            } else if (kb + cj < KDIM) {               // edge-attr region
                const float2 ea = *reinterpret_cast<const float2*>(
                    edge_attr + (size_t)(e0 + lm) * ATTR + (k - EMB));
                au[j] = pack2(ea.x, ea.y);
            } else {                                   // zero padding
                au[j] = 0u;
            }
        }
        const v16bf A = as_bf(au);
        #pragma unroll
        for (int nt = 0; nt < 8; ++nt) {
            const int n = nt * 16 + lm;                // B^T row = Wm row
            v8u bu;
            #pragma unroll
            for (int j = 0; j < 8; ++j) {
                const int cj = (j < 4) ? 2 * j : (16 + 2 * (j - 4));
                const int k  = kb + koff(j, half);
                bu[j] = (kb + cj < KDIM) ? wlds[n * 72 + (k >> 1)] : 0u;
            }
            acc[nt] = __builtin_amdgcn_wmma_f32_16x16x32_bf16(
                false, A, false, as_bf(bu), (short)0, acc[nt], false, false);
        }
    }

    // bias + relu + scatter (D row r -> edge half*8+r, col nt*16+lm)
    #pragma unroll
    for (int nt = 0; nt < 8; ++nt) {
        const int col  = nt * 16 + lm;
        const float bb = bm2[col];
        #pragma unroll
        for (int r = 0; r < 8; ++r) {
            const float x = fmaxf(acc[nt][r] + bb, 0.0f);
            atomicAdd(&agg[(size_t)dstr[r] * EMB + col], x);
        }
    }
}

// -------------------------------------------- node GEMM + LayerNorm ------
__global__ __launch_bounds__(256) void node_gemm_ln(
    const float*    __restrict__ agg,   // [NN,128] f32
    const unsigned* __restrict__ wlb,   // [128,128] bf16 as 64 uints/row
    const float*    __restrict__ bl2,
    const float*    __restrict__ g2,
    const float*    __restrict__ b2,
    float*          __restrict__ out)   // [NN,128]
{
    __shared__ unsigned wlds[8192];     // 32 KB: Wl[2] in bf16
    for (int i = threadIdx.x; i < 8192; i += 256) wlds[i] = wlb[i];
    __syncthreads();

    const int lane = threadIdx.x & 31;
    const int wave = threadIdx.x >> 5;
    const int half = lane >> 4;
    const int lm   = lane & 15;
    const int tile = blockIdx.x * 8 + wave;
    if (tile >= NN / 16) return;
    const int n0   = tile * 16;
    const int arow = n0 + lm;

    v8f acc[8];
    #pragma unroll
    for (int nt = 0; nt < 8; ++nt)
        #pragma unroll
        for (int i = 0; i < 8; ++i) acc[nt][i] = 0.0f;

    #pragma unroll
    for (int kt = 0; kt < 4; ++kt) {    // K = 128 -> 4 tiles of 32
        const int kb = kt * 32;
        v8u au;
        #pragma unroll
        for (int j = 0; j < 8; ++j) {
            const int k = kb + koff(j, half);
            const float2 a2 = *reinterpret_cast<const float2*>(
                agg + (size_t)arow * EMB + k);
            au[j] = pack2(a2.x, a2.y);
        }
        const v16bf A = as_bf(au);
        #pragma unroll
        for (int nt = 0; nt < 8; ++nt) {
            const int n = nt * 16 + lm;
            v8u bu;
            #pragma unroll
            for (int j = 0; j < 8; ++j) {
                const int k = kb + koff(j, half);
                bu[j] = wlds[n * 64 + (k >> 1)];
            }
            acc[nt] = __builtin_amdgcn_wmma_f32_16x16x32_bf16(
                false, A, false, as_bf(bu), (short)0, acc[nt], false, false);
        }
    }

    float bl[8], lg[8], lb[8];
    #pragma unroll
    for (int nt = 0; nt < 8; ++nt) {
        const int col = nt * 16 + lm;
        bl[nt] = bl2[col]; lg[nt] = g2[col]; lb[nt] = b2[col];
    }

    #pragma unroll
    for (int r = 0; r < 8; ++r) {       // one output row per r (per half)
        float xs[8], s = 0.0f, ss = 0.0f;
        #pragma unroll
        for (int nt = 0; nt < 8; ++nt) {
            const float x = acc[nt][r] + bl[nt];
            xs[nt] = x; s += x; ss += x * x;
        }
        #pragma unroll
        for (int m = 1; m < 16; m <<= 1) {   // reduce within 16-lane half
            s  += __shfl_xor(s,  m, 32);
            ss += __shfl_xor(ss, m, 32);
        }
        const float mu   = s * (1.0f / 128.0f);
        const float var  = ss * (1.0f / 128.0f) - mu * mu;
        const float rstd = rsqrtf(var + 1e-5f);
        const int node   = n0 + half * 8 + r;
        #pragma unroll
        for (int nt = 0; nt < 8; ++nt) {
            const int col = nt * 16 + lm;
            const float y = (xs[nt] - mu) * rstd * lg[nt] + lb[nt];
            out[(size_t)node * EMB + col] = fmaxf(y, 0.0f);
        }
    }
}

// ---------------------------------------------------------------- host ----
extern "C" void kernel_launch(void* const* d_in, const int* in_sizes, int n_in,
                              void* d_out, int out_size, void* d_ws, size_t ws_size,
                              hipStream_t stream) {
    const float* nf = (const float*)d_in[0];
    // d_in[1] augmented_view: dead (never feeds the output)
    const int*   ei = (const int*)d_in[2];
    const float* ea = (const float*)d_in[3];
    const float* bc = (const float*)d_in[4];
    const float* Wm = (const float*)d_in[5];
    const float* bm = (const float*)d_in[6];
    const float* Wl = (const float*)d_in[7];
    const float* bl = (const float*)d_in[8];
    const float* lg = (const float*)d_in[9];
    const float* lb = (const float*)d_in[10];
    float* out = (float*)d_out;

    // workspace layout (uints then floats, all 8B-aligned by construction)
    unsigned* nfb = (unsigned*)d_ws;                 // NN*64 uints (bf16 pairs)
    unsigned* wmb = nfb + (size_t)NN * 64;           // 9216 uints
    unsigned* wlb = wmb + 9216;                      // 8192 uints
    float*    agg = (float*)(wlb + 8192);            // NN*128 f32

    const int npNodes = NN * 64;                     // 3.2M pairs
    pack_bf16_pairs<<<(npNodes + 255) / 256, 256, 0, stream>>>(nf, nfb, npNodes);
    pack_bf16_pairs<<<36, 256, 0, stream>>>(Wm + (size_t)LSEL * EMB * KDIM, wmb, 9216);
    pack_bf16_pairs<<<32, 256, 0, stream>>>(Wl + (size_t)LSEL * EMB * EMB,  wlb, 8192);
    init_agg<<<(NN * EMB + 255) / 256, 256, 0, stream>>>(bc, agg, NN * EMB);

    edge_gemm_scatter<<<NE / (16 * 8), 256, 0, stream>>>(
        nfb, wmb, ea, ei, bm + LSEL * EMB, agg);

    node_gemm_ln<<<(NN / 16 + 7) / 8, 256, 0, stream>>>(
        agg, wlb, bl + LSEL * EMB, lg + LSEL * EMB, lb + LSEL * EMB, out);

    (void)in_sizes; (void)n_in; (void)out_size; (void)ws_size;
}
